// UserFactor_15762529976627
// MI455X (gfx1250) — matrix operand order repaired and checked by reading, not compiled, not run on hardware
//
#include <hip/hip_runtime.h>
#include <hip/hip_bf16.h>

typedef unsigned int  u32;
typedef unsigned short u16;
typedef __attribute__((ext_vector_type(16))) __bf16 bf16x16;
typedef __attribute__((ext_vector_type(8)))  float  f32x8;

// ---------------- bf16 helpers ----------------
__device__ __forceinline__ u16 f2bf(float x) {
    u32 u = __float_as_uint(x);
    return (u16)((u + 0x7FFFu + ((u >> 16) & 1u)) >> 16);   // round-to-nearest-even
}
__device__ __forceinline__ float bf2f(u16 h) {
    return __uint_as_float(((u32)h) << 16);
}

// Build a 16-element bf16 fragment from two 8-halfword contiguous LDS chunks.
__device__ __forceinline__ bf16x16 ldfrag(const u16* pa, const u16* pb) {
    union { bf16x16 v; u32 u[8]; } f;
    const u32* A = (const u32*)pa;
    const u32* B = (const u32*)pb;
#pragma unroll
    for (int i = 0; i < 4; ++i) { f.u[i] = A[i]; f.u[4 + i] = B[i]; }
    return f.v;
}

// B-fragment (32x16, bf16) from n-major transposed weight tile wt[n][k].
// Lane: n = nc*16 + (lane&15); elements e<8 -> K = k*32 + (lane>>4)*8 + e, e>=8 -> K = k*32 + 16 + (lane>>4)*8 + (e-8).
__device__ __forceinline__ bf16x16 ldB(const u16* wt, int Kdim, int nc, int k,
                                       int lo, int hi) {
    const u16* p = &wt[(nc * 16 + lo) * Kdim + k * 32 + hi * 8];
    return ldfrag(p, p + 16);
}

#define WAVES_PER_BLOCK 2

// =====================================================================
// Fused gv-MLP (128->64 relu ->64) + atti attention MLP (128->64 relu ->1)
// over N rows. Row input = concat(item_table[idx], rate_table[ridx]).
// Attention input = concat(x_ia, mask * user_table[ctx]).
// Outputs: x bf16 [N][64], attention logit f32 [N].
// =====================================================================
__global__ __launch_bounds__(64) void mlp_gv_atti_kernel(
    const int* __restrict__ pairs,        // [N][2] (item idx, rate idx)
    const int* __restrict__ ctx_ids,      // [N / rows_per_ctx]
    int rows_per_ctx,
    const float* __restrict__ user_table,
    const float* __restrict__ item_table,
    const float* __restrict__ rate_table,
    const float* __restrict__ gvW1, const float* __restrict__ gvb1,   // 128x64, 64
    const float* __restrict__ gvW2, const float* __restrict__ gvb2,   // 64x64, 64
    const float* __restrict__ atW1, const float* __restrict__ atb1,   // 128x64, 64
    const float* __restrict__ atW2, const float* __restrict__ atb2,   // 64x1, 1
    u16*   __restrict__ x_out,            // [N][64] bf16
    float* __restrict__ logit_out,        // [N]
    int numTiles)                         // N/16
{
    __shared__ u16 w1t[64 * 128];   // gv_W1 transposed  [n][k] bf16
    __shared__ u16 w2t[64 * 64];    // gv_W2 transposed
    __shared__ u16 a1t[64 * 128];   // atti_W1 transposed
    __shared__ float b1v[64], b2v[64], a1v[64], a2v[64];
    __shared__ float a2b_s;
    __shared__ u16 in_tile[WAVES_PER_BLOCK][16 * 128];
    __shared__ u16 h_tile[WAVES_PER_BLOCK][16 * 64];

    const int tid = threadIdx.x;
    // ---- preload weights (transpose + convert to bf16) ----
    for (int i = tid; i < 64 * 128; i += blockDim.x) {
        int n = i >> 7, k = i & 127;
        w1t[n * 128 + k] = f2bf(gvW1[k * 64 + n]);
        a1t[n * 128 + k] = f2bf(atW1[k * 64 + n]);
    }
    for (int i = tid; i < 64 * 64; i += blockDim.x) {
        int n = i >> 6, k = i & 63;
        w2t[n * 64 + k] = f2bf(gvW2[k * 64 + n]);
    }
    if (tid < 64) {
        b1v[tid] = gvb1[tid];
        b2v[tid] = gvb2[tid];
        a1v[tid] = atb1[tid];
        a2v[tid] = atW2[tid];
    }
    if (tid == 0) a2b_s = atb2[0];
    __syncthreads();

    const int lane = tid & 31;
    const int wv   = tid >> 5;
    const int lo   = lane & 15;
    const int hi   = lane >> 4;
    u16* itile = in_tile[wv];
    u16* htile = h_tile[wv];

    const int totalWaves = gridDim.x * WAVES_PER_BLOCK;
    const int waveGlobal = blockIdx.x * WAVES_PER_BLOCK + wv;
    const int iters = (numTiles + totalWaves - 1) / totalWaves;

    for (int it = 0; it < iters; ++it) {
        int tileRaw = it * totalWaves + waveGlobal;
        bool valid = tileRaw < numTiles;
        int tile = valid ? tileRaw : (numTiles - 1);

        // ---- stage 1: gather concat(item, rate) -> itile (16x128 bf16) ----
        const int m = lo, h = hi;
        const long g = (long)tile * 16 + m;
        const int idx  = pairs[g * 2 + 0];
        const int ridx = pairs[g * 2 + 1];
        const float* src = h ? (rate_table + (long)ridx * 64)
                             : (item_table + (long)idx * 64);
        u32* drow = (u32*)&itile[m * 128 + h * 64];
#pragma unroll
        for (int j = 0; j < 16; ++j) {
            float4 f = ((const float4*)src)[j];
            drow[2 * j]     = (u32)f2bf(f.x) | ((u32)f2bf(f.y) << 16);
            drow[2 * j + 1] = (u32)f2bf(f.z) | ((u32)f2bf(f.w) << 16);
        }
        __syncthreads();

        // ---- A fragments (16x32 bf16, K-chunks of 128) ----
        bf16x16 A[4];
#pragma unroll
        for (int k = 0; k < 4; ++k) {
            const u16* p = &itile[lo * 128 + k * 32 + hi * 8];
            A[k] = ldfrag(p, p + 16);
        }
        // ---- GEMM1: 128->64, +b1, relu -> htile ----
#pragma unroll
        for (int nc = 0; nc < 4; ++nc) {
            f32x8 acc = {0.f, 0.f, 0.f, 0.f, 0.f, 0.f, 0.f, 0.f};
#pragma unroll
            for (int k = 0; k < 4; ++k)
                acc = __builtin_amdgcn_wmma_f32_16x16x32_bf16(
                    false, A[k], false, ldB(w1t, 128, nc, k, lo, hi),
                    (short)0, acc, false, false);
            float bb = b1v[nc * 16 + lo];
#pragma unroll
            for (int r = 0; r < 8; ++r)
                htile[(r + 8 * hi) * 64 + nc * 16 + lo] =
                    f2bf(fmaxf(acc[r] + bb, 0.f));
        }
        __syncthreads();

        // ---- GEMM2: 64->64, +b2 (no relu) -> x into itile cols 0..63 ----
        bf16x16 H[2];
#pragma unroll
        for (int k = 0; k < 2; ++k) {
            const u16* p = &htile[lo * 64 + k * 32 + hi * 8];
            H[k] = ldfrag(p, p + 16);
        }
#pragma unroll
        for (int nc = 0; nc < 4; ++nc) {
            f32x8 acc = {0.f, 0.f, 0.f, 0.f, 0.f, 0.f, 0.f, 0.f};
#pragma unroll
            for (int k = 0; k < 2; ++k)
                acc = __builtin_amdgcn_wmma_f32_16x16x32_bf16(
                    false, H[k], false, ldB(w2t, 64, nc, k, lo, hi),
                    (short)0, acc, false, false);
            float bb = b2v[nc * 16 + lo];
#pragma unroll
            for (int r = 0; r < 8; ++r)
                itile[(r + 8 * hi) * 128 + nc * 16 + lo] = f2bf(acc[r] + bb);
        }
        __syncthreads();

        // ---- store x_ia to global; gather masked user ctx into cols 64..127 ----
        if (valid) {
            u32* dst = (u32*)&x_out[(size_t)g * 64 + h * 32];
            const u32* s = (const u32*)&itile[m * 128 + h * 32];
#pragma unroll
            for (int j = 0; j < 16; ++j) dst[j] = s[j];
        }
        const int cu = ctx_ids[g / rows_per_ctx];
        const float msk = (idx > 0) ? 1.f : 0.f;
        const float* csrc = user_table + (long)cu * 64 + h * 32;
        u32* crow = (u32*)&itile[m * 128 + 64 + h * 32];
#pragma unroll
        for (int j = 0; j < 8; ++j) {
            float4 f = ((const float4*)csrc)[j];
            crow[2 * j]     = (u32)f2bf(f.x * msk) | ((u32)f2bf(f.y * msk) << 16);
            crow[2 * j + 1] = (u32)f2bf(f.z * msk) | ((u32)f2bf(f.w * msk) << 16);
        }
        __syncthreads();

        // ---- atti GEMM: 128->64, +a1, relu -> htile ----
#pragma unroll
        for (int k = 0; k < 4; ++k) {
            const u16* p = &itile[lo * 128 + k * 32 + hi * 8];
            A[k] = ldfrag(p, p + 16);
        }
#pragma unroll
        for (int nc = 0; nc < 4; ++nc) {
            f32x8 acc = {0.f, 0.f, 0.f, 0.f, 0.f, 0.f, 0.f, 0.f};
#pragma unroll
            for (int k = 0; k < 4; ++k)
                acc = __builtin_amdgcn_wmma_f32_16x16x32_bf16(
                    false, A[k], false, ldB(a1t, 128, nc, k, lo, hi),
                    (short)0, acc, false, false);
            float bb = a1v[nc * 16 + lo];
#pragma unroll
            for (int r = 0; r < 8; ++r)
                htile[(r + 8 * hi) * 64 + nc * 16 + lo] =
                    f2bf(fmaxf(acc[r] + bb, 0.f));
        }
        __syncthreads();

        // ---- logit: dot(hidden_row, atW2) + atb2 (half-lane split + shfl) ----
        float sacc = 0.f;
#pragma unroll
        for (int j = 0; j < 32; ++j) {
            int c = h * 32 + j;
            sacc += bf2f(htile[lo * 64 + c]) * a2v[c];
        }
        sacc += __shfl_xor(sacc, 16, 32);
        if (h == 0 && valid)
            logit_out[(size_t)tile * 16 + lo] = sacc + a2b_s;
        __syncthreads();
    }
}

// =====================================================================
// Masked softmax over L entries per group + weighted sum + 64x64 matvec + relu
// =====================================================================
template <bool XBF16>
__global__ __launch_bounds__(64) void agg_kernel(
    const void* __restrict__ xin,          // [G*L][64] bf16 or f32
    const float* __restrict__ logits,      // [G*L]
    const int* __restrict__ mask_idx,      // [G*L*mask_stride]
    int mask_stride,
    const float* __restrict__ W,           // 64x64 [k][n]
    const float* __restrict__ bias,        // 64
    float* __restrict__ out,               // [G][64]
    int L)
{
    __shared__ float e[128];
    __shared__ float s[64];
    __shared__ float red[64];
    const int g = blockIdx.x;
    const int t = threadIdx.x;

    float partial = 0.f;
    for (int l = t; l < L; l += 64) {
        float lg = logits[(size_t)g * L + l];
        int mi = mask_idx[((size_t)g * L + l) * mask_stride];
        float el = (mi > 0) ? __expf(lg) : 0.f;
        e[l] = el;
        partial += el;
    }
    red[t] = partial;
    __syncthreads();
    for (int o = 32; o >= 1; o >>= 1) {
        if (t < o) red[t] += red[t + o];
        __syncthreads();
    }
    const float inv = 1.f / (red[0] + 1e-10f);

    float sd = 0.f;
    if (XBF16) {
        const u16* xb = (const u16*)xin;
        for (int l = 0; l < L; ++l)
            sd += e[l] * bf2f(xb[((size_t)g * L + l) * 64 + t]);
    } else {
        const float* xf = (const float*)xin;
        for (int l = 0; l < L; ++l)
            sd += e[l] * xf[((size_t)g * L + l) * 64 + t];
    }
    s[t] = sd * inv;
    __syncthreads();

    float acc = bias[t];
    for (int k = 0; k < 64; ++k) acc += s[k] * W[k * 64 + t];
    out[(size_t)g * 64 + t] = fmaxf(acc, 0.f);
}

// =====================================================================
// beta logits: MLP(concat(h_oI, user_table[nbr])) 128->64 relu ->1
// =====================================================================
__global__ __launch_bounds__(64) void beta_kernel(
    const float* __restrict__ h_oI,        // [G][64]
    const int* __restrict__ u_user_pad,    // [G]
    const float* __restrict__ user_table,
    const float* __restrict__ W1, const float* __restrict__ b1,
    const float* __restrict__ W2, const float* __restrict__ b2,
    float* __restrict__ beta_out)
{
    __shared__ float in[128];
    __shared__ float red[64];
    const int row = blockIdx.x;
    const int t = threadIdx.x;
    const int nbr = u_user_pad[row];
    in[t]      = h_oI[(size_t)row * 64 + t];
    in[64 + t] = user_table[(size_t)nbr * 64 + t];
    __syncthreads();
    float a = b1[t];
    for (int k = 0; k < 128; ++k) a += in[k] * W1[k * 64 + t];
    red[t] = fmaxf(a, 0.f) * W2[t];
    __syncthreads();
    for (int o = 32; o >= 1; o >>= 1) {
        if (t < o) red[t] += red[t + o];
        __syncthreads();
    }
    if (t == 0) beta_out[row] = red[0] + b2[0];
}

// =====================================================================
// final combine: relu chain 128->64->64->64
// =====================================================================
__global__ __launch_bounds__(64) void final_kernel(
    const float* __restrict__ h_iI, const float* __restrict__ h_iS,
    const float* __restrict__ W1, const float* __restrict__ b1,
    const float* __restrict__ W2, const float* __restrict__ b2,
    const float* __restrict__ W3, const float* __restrict__ b3,
    float* __restrict__ out)
{
    __shared__ float in[128];
    __shared__ float h1[64];
    __shared__ float h2[64];
    const int b = blockIdx.x;
    const int t = threadIdx.x;
    in[t]      = h_iI[(size_t)b * 64 + t];
    in[64 + t] = h_iS[(size_t)b * 64 + t];
    __syncthreads();
    float a = b1[t];
    for (int k = 0; k < 128; ++k) a += in[k] * W1[k * 64 + t];
    h1[t] = fmaxf(a, 0.f);
    __syncthreads();
    a = b2[t];
    for (int k = 0; k < 64; ++k) a += h1[k] * W2[k * 64 + t];
    h2[t] = fmaxf(a, 0.f);
    __syncthreads();
    a = b3[t];
    for (int k = 0; k < 64; ++k) a += h2[k] * W3[k * 64 + t];
    out[(size_t)b * 64 + t] = fmaxf(a, 0.f);
}

// =====================================================================
extern "C" void kernel_launch(void* const* d_in, const int* in_sizes, int n_in,
                              void* d_out, int out_size, void* d_ws, size_t ws_size,
                              hipStream_t stream) {
    (void)in_sizes; (void)n_in; (void)out_size; (void)ws_size;
    const int*   uids            = (const int*)d_in[0];
    const int*   u_item_pad      = (const int*)d_in[1];
    const int*   u_user_pad      = (const int*)d_in[2];
    const int*   u_user_item_pad = (const int*)d_in[3];
    const float* user_table      = (const float*)d_in[4];
    const float* item_table      = (const float*)d_in[5];
    const float* rate_table      = (const float*)d_in[6];
    const float* gv_W1  = (const float*)d_in[7];
    const float* gv_b1  = (const float*)d_in[8];
    const float* gv_W2  = (const float*)d_in[9];
    const float* gv_b2  = (const float*)d_in[10];
    const float* atti_W1 = (const float*)d_in[11];
    const float* atti_b1 = (const float*)d_in[12];
    const float* atti_W2 = (const float*)d_in[13];
    const float* atti_b2 = (const float*)d_in[14];
    const float* agg_W  = (const float*)d_in[15];
    const float* agg_b  = (const float*)d_in[16];
    const float* attu_W1 = (const float*)d_in[17];
    const float* attu_b1 = (const float*)d_in[18];
    const float* attu_W2 = (const float*)d_in[19];
    const float* attu_b2 = (const float*)d_in[20];
    const float* aggn_W = (const float*)d_in[21];
    const float* aggn_b = (const float*)d_in[22];
    const float* cm_W1  = (const float*)d_in[23];
    const float* cm_b1  = (const float*)d_in[24];
    const float* cm_W2  = (const float*)d_in[25];
    const float* cm_b2  = (const float*)d_in[26];
    const float* cm_W3  = (const float*)d_in[27];
    const float* cm_b3  = (const float*)d_in[28];
    float* out = (float*)d_out;

    const int B = 512, P = 100, Q = 40, P2 = 40;
    const size_t N1 = (size_t)B * P;        // 51200 direct rows
    const size_t N2 = (size_t)B * Q * P2;   // 819200 social rows
    const size_t G2 = (size_t)B * Q;        // 20480

    size_t off = 0;
    auto carve = [&](size_t bytes) -> void* {
        void* p = (char*)d_ws + off;
        off += (bytes + 255) & ~(size_t)255;
        return p;
    };
    u16*   x_s     = (u16*)carve(N2 * 64 * sizeof(u16));
    float* logit_s = (float*)carve(N2 * sizeof(float));
    u16*   x_d     = (u16*)carve(N1 * 64 * sizeof(u16));
    float* logit_d = (float*)carve(N1 * sizeof(float));
    float* h_oI    = (float*)carve(G2 * 64 * sizeof(float));
    float* h_iI    = (float*)carve((size_t)B * 64 * sizeof(float));
    float* beta    = (float*)carve(G2 * sizeof(float));
    float* h_iS    = (float*)carve((size_t)B * 64 * sizeof(float));

    // Social branch: x_ia_s + alpha_s logits (819200 rows, 51200 tiles)
    mlp_gv_atti_kernel<<<1024, 64, 0, stream>>>(
        u_user_item_pad, u_user_pad, P2,
        user_table, item_table, rate_table,
        gv_W1, gv_b1, gv_W2, gv_b2,
        atti_W1, atti_b1, atti_W2, atti_b2,
        x_s, logit_s, (int)(N2 / 16));

    // Direct branch: x_ia + alpha logits (51200 rows, 3200 tiles)
    mlp_gv_atti_kernel<<<400, 64, 0, stream>>>(
        u_item_pad, uids, P,
        user_table, item_table, rate_table,
        gv_W1, gv_b1, gv_W2, gv_b2,
        atti_W1, atti_b1, atti_W2, atti_b2,
        x_d, logit_d, (int)(N1 / 16));

    // h_oI: softmax over P2, aggregate, agg matmul + relu
    agg_kernel<true><<<(int)G2, 64, 0, stream>>>(
        x_s, logit_s, u_user_item_pad, 2, agg_W, agg_b, h_oI, P2);

    // h_iI: softmax over P, aggregate, agg matmul + relu
    agg_kernel<true><<<B, 64, 0, stream>>>(
        x_d, logit_d, u_item_pad, 2, agg_W, agg_b, h_iI, P);

    // beta logits over (b,q)
    beta_kernel<<<(int)G2, 64, 0, stream>>>(
        h_oI, u_user_pad, user_table,
        attu_W1, attu_b1, attu_W2, attu_b2, beta);

    // h_iS: softmax over Q, aggregate h_oI, aggn matmul + relu
    agg_kernel<false><<<B, 64, 0, stream>>>(
        h_oI, beta, u_user_pad, 1, aggn_W, aggn_b, h_iS, Q);

    // final combine MLP
    final_kernel<<<B, 64, 0, stream>>>(
        h_iI, h_iS, cm_W1, cm_b1, cm_W2, cm_b2, cm_W3, cm_b3, out);
}
